// QEffPlamoAttention_15539191677652
// MI455X (gfx1250) — matrix operand order repaired and checked by reading, not compiled, not run on hardware
//
#include <hip/hip_runtime.h>
#include <hip/hip_bf16.h>
#include <cstdint>
#include <cstddef>
#include <type_traits>

// ---------------------------------------------------------------------------
// PLaMo GQA attention block for MI455X (gfx1250), wave32 + WMMA f16.
//   X[3072,2048] -> Q/K/V proj -> RoPE -> causal flash attention -> O proj.
// All GEMMs use v_wmma_f32_16x16x32_f16 (16x16 tiles, K-step 32).
// ---------------------------------------------------------------------------

#define S_LEN   3072
#define HID     2048
#define NHQ     16
#define NHK     2
#define HD      128
#define KVDIM   (NHK * HD)      // 256
#define THETA   10000.0f
#define ASCALE  0.08838834764831845f   // 1/sqrt(128), folded into Q at RoPE

typedef __attribute__((ext_vector_type(16))) _Float16 v16h;
typedef __attribute__((ext_vector_type(8)))  _Float16 v8h;
typedef __attribute__((ext_vector_type(8)))  float    v8f;

static __device__ __forceinline__ v16h cat8(v8h lo, v8h hi) {
  return __builtin_shufflevector(lo, hi, 0,1,2,3,4,5,6,7,8,9,10,11,12,13,14,15);
}
static __device__ __forceinline__ v8h ldh8(const _Float16* p) {
  return *(const v8h*)p;   // 16B-aligned by construction -> global_load_b128
}
static __device__ __forceinline__ v8f wmma16(v16h a, v16h b, v8f c) {
  // (neg_a, A, neg_b, B, c_mod, C, reuse_a, reuse_b)
  return __builtin_amdgcn_wmma_f32_16x16x32_f16(false, a, false, b, (short)0, c,
                                                false, false);
}

// ---------------------------------------------------------------------------
// fp32 -> f16 convert
__global__ void cvt_f16_kernel(const float* __restrict__ in,
                               _Float16* __restrict__ out, int n) {
  int i = blockIdx.x * 256 + threadIdx.x;
  if (i < n) out[i] = (_Float16)in[i];
}

// W[Nrows][Kcols] fp32 (row-major, torch Linear weight) -> Wt[k][n] f16
// so that WMMA B-fragments (lane = n, element = k) are contiguous loads.
__global__ void transpose_f16_kernel(const float* __restrict__ W,
                                     _Float16* __restrict__ Wt,
                                     int Nrows, int Kcols) {
  int k = blockIdx.x * 256 + threadIdx.x;
  int n = blockIdx.y;
  if (k < Kcols)
    Wt[(size_t)k * Nrows + n] = (_Float16)W[(size_t)n * Kcols + k];
}

// ---------------------------------------------------------------------------
// Generic WMMA GEMM:  C[M][N] = A[M][K] * Bt[K][N]
// A row-major f16, Bt K-major f16. One wave per 64x64 tile, 4x4 accumulators.
template <bool OUTF32>
__global__ __launch_bounds__(32, 1)
void gemm_wmma_kernel(const _Float16* __restrict__ A,
                      const _Float16* __restrict__ Bt,
                      float* __restrict__ Cf,
                      _Float16* __restrict__ Ch,
                      int K, int lda, int ldb, int ldc) {
  const int lane = threadIdx.x;
  const int rlo  = lane & 15;       // row (A) / col (B,C) selector
  const int rhi  = lane >> 4;       // K-chunk selector for A frags
  const int mbase = blockIdx.y * 64;
  const int nbase = blockIdx.x * 64;

  v8f acc[4][4] = {};

  const _Float16* arow = A + (size_t)(mbase + rlo) * lda + rhi * 8;
  const _Float16* brow = Bt + (size_t)(rhi * 8) * ldb + nbase + rlo;

  for (int kk = 0; kk < K; kk += 32) {
    v16h af[4], bf[4];
#pragma unroll
    for (int mt = 0; mt < 4; ++mt) {
      const _Float16* p = arow + (size_t)(mt * 16) * lda + kk;
      af[mt] = cat8(ldh8(p), ldh8(p + 16));
    }
#pragma unroll
    for (int nt = 0; nt < 4; ++nt) {
      const _Float16* p = brow + (size_t)kk * ldb + nt * 16;
      for (int s = 0; s < 16; ++s)
        bf[nt][s] = p[(size_t)((s & 7) + (s >> 3) * 16) * ldb];
    }
#pragma unroll
    for (int mt = 0; mt < 4; ++mt)
#pragma unroll
      for (int nt = 0; nt < 4; ++nt)
        acc[mt][nt] = wmma16(af[mt], bf[nt], acc[mt][nt]);
  }

#pragma unroll
  for (int mt = 0; mt < 4; ++mt)
#pragma unroll
    for (int nt = 0; nt < 4; ++nt)
#pragma unroll
      for (int r = 0; r < 8; ++r) {
        int row = mbase + mt * 16 + rhi * 8 + r;
        int col = nbase + nt * 16 + rlo;
        if constexpr (OUTF32)
          Cf[(size_t)row * ldc + col] = acc[mt][nt][r];
        else
          Ch[(size_t)row * ldc + col] = (_Float16)acc[mt][nt][r];
      }
}

// ---------------------------------------------------------------------------
// RoPE on Q (row-major in/out, [S][HQ*128]); folds in the 1/sqrt(D) scale.
__global__ void rope_q_kernel(const _Float16* __restrict__ Qp,
                              _Float16* __restrict__ Qr,
                              const int* __restrict__ pos_ids) {
  int idx = blockIdx.x * 256 + threadIdx.x;       // over S*HQ*HD
  if (idx >= S_LEN * NHQ * HD) return;
  int s = idx / (NHQ * HD);
  int d = idx & (HD - 1);
  float freq = __powf(THETA, -(float)(d & 63) * (1.0f / 64.0f));
  float sn, cs;
  __sincosf((float)pos_ids[s] * freq, &sn, &cs);
  float x = (float)Qp[idx];
  float xr = (float)Qp[(d < 64) ? idx + 64 : idx - 64];
  if (d < 64) xr = -xr;
  Qr[idx] = (_Float16)((x * cs + xr * sn) * ASCALE);
}

// RoPE on K + transpose: Kp[S][HK*128] -> Kt[(h*128+d)][S]  (B-frag friendly)
__global__ void rope_k_kernel(const _Float16* __restrict__ Kp,
                              _Float16* __restrict__ Kt,
                              const int* __restrict__ pos_ids) {
  int idx = blockIdx.x * 256 + threadIdx.x;       // over S*HK*HD
  if (idx >= S_LEN * NHK * HD) return;
  int s   = idx / (NHK * HD);
  int rem = idx - s * (NHK * HD);
  int h   = rem >> 7;
  int d   = rem & (HD - 1);
  float freq = __powf(THETA, -(float)(d & 63) * (1.0f / 64.0f));
  float sn, cs;
  __sincosf((float)pos_ids[s] * freq, &sn, &cs);
  float x = (float)Kp[idx];
  float xr = (float)Kp[(d < 64) ? idx + 64 : idx - 64];
  if (d < 64) xr = -xr;
  Kt[(size_t)(h * HD + d) * S_LEN + s] = (_Float16)(x * cs + xr * sn);
}

// ---------------------------------------------------------------------------
// Flash attention: one wave per (16-row q-block, head). Causal, GQA (h%2).
// Main loop is mask-free; only the diagonal-straddling tail block masks.
__global__ __launch_bounds__(32, 1)
void attn_kernel(const _Float16* __restrict__ Qr,
                 const _Float16* __restrict__ Kt,
                 const _Float16* __restrict__ V,
                 _Float16* __restrict__ O) {
  __shared__ __align__(16) _Float16 Plds[16 * 32];

  const int lane  = threadIdx.x;
  const int rlo   = lane & 15;
  const int rhi   = lane >> 4;
  const int qbase = blockIdx.x * 16;
  const int h     = blockIdx.y;
  const int hkv   = h & (NHK - 1);           // tile([k0,k1],8)[:16] -> h % 2

  // Q A-fragments for the 16x128 row block (4 K-steps of 32), pre-scaled
  v16h qf[4];
  {
    const _Float16* qrow = Qr + (size_t)(qbase + rlo) * HID + h * HD;
#pragma unroll
    for (int ks = 0; ks < 4; ++ks) {
      const _Float16* p = qrow + ks * 32 + rhi * 8;
      qf[ks] = cat8(ldh8(p), ldh8(p + 16));
    }
  }

  v8f oacc[8] = {};
  float mrow[8], lrow[8];
#pragma unroll
  for (int r = 0; r < 8; ++r) { mrow[r] = -1e30f; lrow[r] = 0.0f; }

  const _Float16* ktb = Kt + (size_t)(hkv * HD) * S_LEN;
  const _Float16* vtb = V + hkv * HD;

  auto step = [&](int kb, auto domask_c) {
    constexpr bool DOMASK = decltype(domask_c)::value;
    // ---- scores: two 16x16 tiles over 32 keys, K-dim = head dim (128)
    v8f s0 = {}, s1 = {};
#pragma unroll
    for (int ks = 0; ks < 4; ++ks) {
      const _Float16* kp = ktb + (size_t)(ks * 32 + rhi * 8) * S_LEN + kb + rlo;
      v16h b0, b1;
      for (int s = 0; s < 16; ++s) {
        size_t o = (size_t)((s & 7) + (s >> 3) * 16) * S_LEN;
        b0[s] = kp[o];
        b1[s] = kp[o + 16];
      }
      s0 = wmma16(qf[ks], b0, s0);
      s1 = wmma16(qf[ks], b1, s1);
    }

    // ---- online softmax; per-row: reduce, rescale O, emit P into LDS
#pragma unroll
    for (int r = 0; r < 8; ++r) {
      float a0 = s0[r];
      float a1 = s1[r];
      if constexpr (DOMASK) {
        int row = qbase + rhi * 8 + r;
        if (kb + rlo > row)      a0 = -10000.0f;  // reference: where(mask,-1e4,s)
        if (kb + 16 + rlo > row) a1 = -10000.0f;
      }
      float t = fmaxf(a0, a1);
      t = fmaxf(t, __shfl_xor(t, 1, 32));
      t = fmaxf(t, __shfl_xor(t, 2, 32));
      t = fmaxf(t, __shfl_xor(t, 4, 32));
      t = fmaxf(t, __shfl_xor(t, 8, 32));
      float mnew = fmaxf(mrow[r], t);
      float corr = __expf(mrow[r] - mnew);
      float p0 = __expf(a0 - mnew);
      float p1 = __expf(a1 - mnew);
      float su = p0 + p1;
      su += __shfl_xor(su, 1, 32);
      su += __shfl_xor(su, 2, 32);
      su += __shfl_xor(su, 4, 32);
      su += __shfl_xor(su, 8, 32);
      lrow[r] = lrow[r] * corr + su;
      mrow[r] = mnew;
      int prow = rhi * 8 + r;
      Plds[prow * 32 + rlo]      = (_Float16)p0;
      Plds[prow * 32 + 16 + rlo] = (_Float16)p1;
#pragma unroll
      for (int t8 = 0; t8 < 8; ++t8) oacc[t8][r] = oacc[t8][r] * corr;
    }

    // ---- transpose P (C-layout) -> A-layout via LDS read
    const _Float16* pr = &Plds[rlo * 32 + rhi * 8];
    v16h pf = cat8(*(const v8h*)pr, *(const v8h*)(pr + 16));

    // ---- O += P @ V  (32 keys, 8 d-tiles of 16)
    const _Float16* vrow = vtb + (size_t)(kb + rhi * 8) * KVDIM + rlo;
#pragma unroll
    for (int t = 0; t < 8; ++t) {
      v16h vf;
      for (int s = 0; s < 16; ++s)
        vf[s] = vrow[(size_t)((s & 7) + (s >> 3) * 16) * KVDIM + t * 16];
      oacc[t] = wmma16(pf, vf, oacc[t]);
    }
  };

  // last key-block start: only block that straddles the causal diagonal
  const int kbL = ((qbase + 16 + 31) / 32 - 1) * 32;
  for (int kb = 0; kb < kbL; kb += 32)
    step(kb, std::integral_constant<bool, false>{});
  step(kbL, std::integral_constant<bool, true>{});

  // ---- finalize: divide by l, store f16 row-major [S][HQ*HD]
#pragma unroll
  for (int r = 0; r < 8; ++r) {
    float inv = 1.0f / lrow[r];
    int row = qbase + rhi * 8 + r;
#pragma unroll
    for (int t = 0; t < 8; ++t)
      O[(size_t)row * HID + h * HD + t * 16 + rlo] =
          (_Float16)(oacc[t][r] * inv);
  }
}

// ---------------------------------------------------------------------------
extern "C" void kernel_launch(void* const* d_in, const int* in_sizes, int n_in,
                              void* d_out, int out_size, void* d_ws, size_t ws_size,
                              hipStream_t stream) {
  (void)in_sizes; (void)n_in; (void)out_size; (void)ws_size;
  const float* hs  = (const float*)d_in[0];
  // d_in[1] = attention_mask (causal; reproduced analytically in-kernel)
  const int*   pos = (const int*)d_in[2];
  const float* Wq  = (const float*)d_in[3];
  const float* Wk  = (const float*)d_in[4];
  const float* Wv  = (const float*)d_in[5];
  const float* Wo  = (const float*)d_in[6];
  float* out = (float*)d_out;

  char* ws = (char*)d_ws;
  size_t off = 0;
  auto carve = [&](size_t bytes) -> void* {
    void* p = ws + off;
    off = (off + bytes + 255) & ~(size_t)255;
    return p;
  };
  _Float16* X16 = (_Float16*)carve((size_t)S_LEN * HID * 2);
  _Float16* WqT = (_Float16*)carve((size_t)HID * HID * 2);
  _Float16* WkT = (_Float16*)carve((size_t)HID * KVDIM * 2);
  _Float16* WvT = (_Float16*)carve((size_t)HID * KVDIM * 2);
  _Float16* WoT = (_Float16*)carve((size_t)HID * HID * 2);
  _Float16* Qp  = (_Float16*)carve((size_t)S_LEN * HID * 2);
  _Float16* Kp  = (_Float16*)carve((size_t)S_LEN * KVDIM * 2);
  _Float16* Vp  = (_Float16*)carve((size_t)S_LEN * KVDIM * 2);
  _Float16* Qr  = (_Float16*)carve((size_t)S_LEN * HID * 2);
  _Float16* Ktr = (_Float16*)carve((size_t)KVDIM * S_LEN * 2);
  _Float16* Att = (_Float16*)carve((size_t)S_LEN * HID * 2);

  // 1) precision conversion / weight transposes
  {
    int n = S_LEN * HID;
    cvt_f16_kernel<<<(n + 255) / 256, 256, 0, stream>>>(hs, X16, n);
  }
  transpose_f16_kernel<<<dim3(HID / 256, HID),   256, 0, stream>>>(Wq, WqT, HID,   HID);
  transpose_f16_kernel<<<dim3(HID / 256, KVDIM), 256, 0, stream>>>(Wk, WkT, KVDIM, HID);
  transpose_f16_kernel<<<dim3(HID / 256, KVDIM), 256, 0, stream>>>(Wv, WvT, KVDIM, HID);
  transpose_f16_kernel<<<dim3(HID / 256, HID),   256, 0, stream>>>(Wo, WoT, HID,   HID);

  // 2) QKV projections (WMMA GEMMs, f16 out)
  gemm_wmma_kernel<false><<<dim3(HID / 64,   S_LEN / 64), 32, 0, stream>>>(
      X16, WqT, nullptr, Qp, HID, HID, HID, HID);
  gemm_wmma_kernel<false><<<dim3(KVDIM / 64, S_LEN / 64), 32, 0, stream>>>(
      X16, WkT, nullptr, Kp, HID, HID, KVDIM, KVDIM);
  gemm_wmma_kernel<false><<<dim3(KVDIM / 64, S_LEN / 64), 32, 0, stream>>>(
      X16, WvT, nullptr, Vp, HID, HID, KVDIM, KVDIM);

  // 3) RoPE (K also transposed to [d][s] for score B-fragments)
  {
    int nq = S_LEN * NHQ * HD;
    rope_q_kernel<<<(nq + 255) / 256, 256, 0, stream>>>(Qp, Qr, pos);
    int nk = S_LEN * NHK * HD;
    rope_k_kernel<<<(nk + 255) / 256, 256, 0, stream>>>(Kp, Ktr, pos);
  }

  // 4) causal flash attention
  attn_kernel<<<dim3(S_LEN / 16, NHQ), 32, 0, stream>>>(Qr, Ktr, Vp, Att);

  // 5) output projection (fp32 out)
  gemm_wmma_kernel<true><<<dim3(HID / 64, S_LEN / 64), 32, 0, stream>>>(
      Att, WoT, out, nullptr, HID, HID, HID, HID);
}
